// SelfBiLayer_45818711114069
// MI455X (gfx1250) — compile-verified, matrix operand order checked by reading, not run on hardware
//
#include <hip/hip_runtime.h>

#define B_  256
#define L_  2048
#define E_  64
#define DA_ 64
#define R_  32

typedef __attribute__((ext_vector_type(16))) __bf16 v16bf;
typedef __attribute__((ext_vector_type(8)))  float  v8f;

union Frag { v16bf bf; unsigned u[8]; };

// ---- f32 -> bf16 conversions via CDNA5 V_FMA_MIXLO/MIXHI_BF16 (VOP3P op 62/63) ----
// One VALU op per converted value; packs two bf16 into one dword for LDS stores.

// pack bf16(a), bf16(b) into one dword
__device__ inline unsigned pk2bf(float a, float b) {
  unsigned r;
  asm("v_fma_mixlo_bf16 %0, %1, 1.0, 0\n\t"
      "v_fma_mixhi_bf16 %0, %2, 1.0, 0"
      : "=&v"(r) : "v"(a), "v"(b));
  return r;
}

// pack bf16(a*s), bf16(b*s) into one dword (mask multiply folded into the fma)
__device__ inline unsigned pk2bf_scaled(float a, float b, float s) {
  unsigned r;
  asm("v_fma_mixlo_bf16 %0, %1, %3, 0\n\t"
      "v_fma_mixhi_bf16 %0, %2, %3, 0"
      : "=&v"(r) : "v"(a), "v"(b), "v"(s));
  return r;
}

__device__ inline unsigned short f2bf(float f) {
  unsigned r;
  asm("v_fma_mixlo_bf16 %0, %1, 1.0, 0" : "=v"(r) : "v"(f));
  return (unsigned short)r;
}

// ---- hardware tanh (V_TANH_F32 on gfx1250) ----
__device__ inline float fast_tanh(float x) {
#if __has_builtin(__builtin_amdgcn_tanhf)
  return __builtin_amdgcn_tanhf(x);
#else
  return tanhf(x);
#endif
}

// WMMA A-fragment (16x32 bf16) from a row-major bf16 matrix (leading dim ld).
// Also used for B-fragments when the stored matrix is Bᵀ (row = n, contiguous k).
// ISA layout: lanes 0-15: K=0-7 (v0-3), K=16-23 (v4-7); lanes 16-31: +8. Pairs per dword.
__device__ inline Frag load_frag_rowmajor(const unsigned short* base, int ld,
                                          int rowOff, int kOff, int lane) {
  Frag f;
  int m  = rowOff + (lane & 15);
  int kb = kOff + ((lane >> 4) << 3);
  const unsigned* row = (const unsigned*)(base + m * ld);
#pragma unroll
  for (int i = 0; i < 4; ++i) {
    f.u[i]     = row[(kb >> 1) + i];
    f.u[4 + i] = row[((kb + 16) >> 1) + i];
  }
  return f;
}

// WMMA B-fragment (32x16 bf16) from row-major [K][N] bf16 matrix (ldN elems/row).
__device__ inline Frag load_fragB_colstride(const unsigned short* base, int ldN,
                                            int kOff, int nOff, int lane) {
  Frag f;
  int n  = nOff + (lane & 15);
  int kb = kOff + ((lane >> 4) << 3);
  const unsigned short* col = base + n;
#pragma unroll
  for (int i = 0; i < 4; ++i) {
    unsigned lo  = col[(kb + 2 * i)      * ldN];
    unsigned hi  = col[(kb + 2 * i + 1)  * ldN];
    unsigned lo2 = col[(kb + 16 + 2 * i) * ldN];
    unsigned hi2 = col[(kb + 17 + 2 * i) * ldN];
    f.u[i]     = lo  | (hi  << 16);
    f.u[4 + i] = lo2 | (hi2 << 16);
  }
  return f;
}

__device__ inline v8f wmma_bf16(const Frag& a, const Frag& b, v8f c) {
  return __builtin_amdgcn_wmma_f32_16x16x32_bf16(false, a.bf, false, b.bf,
                                                 (short)0, c, false, false);
}

__global__ __launch_bounds__(128)
void SelfBiLayer_kernel(const float* __restrict__ x, const float* __restrict__ mask,
                        const float* __restrict__ W1, const float* __restrict__ W2,
                        float* __restrict__ out) {
  __shared__ unsigned short sW1[DA_ * E_];          // bf16 W1 [d][e]
  __shared__ unsigned short sW2[R_ * DA_];          // bf16 W2 [r][d]
  __shared__ unsigned       sXu[4][32 * (E_ / 2)];  // per-wave bf16 xm tile [l][e] (packed)
  __shared__ unsigned short sHT[4][32 * DA_];       // per-wave bf16 H transposed [l][d]
  __shared__ unsigned short sA [4][R_ * 32];        // per-wave bf16 exp scores [r][l]
  __shared__ float          sMf[4][32];             // per-wave token masks
  __shared__ float          sS [4][R_];             // per-wave partial row sums
  __shared__ float          sP [4][R_ * E_];        // per-wave partial pooled output

  const int b    = blockIdx.x;
  const int tid  = threadIdx.x;
  const int wv   = tid >> 5;
  const int lane = tid & 31;
  const int g    = lane >> 4;

  for (int i = tid; i < DA_ * E_; i += 128) sW1[i] = f2bf(W1[i]);
  for (int i = tid; i < R_ * DA_; i += 128) sW2[i] = f2bf(W2[i]);
  __syncthreads();

  // Persistent weight fragments: W1 4 M-tiles x 2 K-steps, W2 2 x 2.
  Frag w1f[4][2], w2f[2][2];
#pragma unroll
  for (int md = 0; md < 4; ++md)
#pragma unroll
    for (int ks = 0; ks < 2; ++ks)
      w1f[md][ks] = load_frag_rowmajor(sW1, E_, 16 * md, 32 * ks, lane);
#pragma unroll
  for (int mr = 0; mr < 2; ++mr)
#pragma unroll
    for (int ks = 0; ks < 2; ++ks)
      w2f[mr][ks] = load_frag_rowmajor(sW2, DA_, 16 * mr, 32 * ks, lane);

  v8f P[2][4];
#pragma unroll
  for (int mt = 0; mt < 2; ++mt)
#pragma unroll
    for (int nt = 0; nt < 4; ++nt)
      P[mt][nt] = v8f{0.f,0.f,0.f,0.f,0.f,0.f,0.f,0.f};

  float Sacc[2][8];
#pragma unroll
  for (int mt = 0; mt < 2; ++mt)
#pragma unroll
    for (int v = 0; v < 8; ++v) Sacc[mt][v] = 0.f;

  unsigned short* sX16 = (unsigned short*)sXu[wv];

  for (int it = 0; it < L_ / 128; ++it) {
    // ---- load this wave's 32-token tile: HBM -> masked bf16 in LDS ----
    const int tok = it * 128 + wv * 32 + lane;
    const float mk = mask[(size_t)b * L_ + tok];
    sMf[wv][lane] = mk;
    const float4* xr = (const float4*)(x + ((size_t)b * L_ + tok) * E_);
    unsigned* dst = &sXu[wv][lane * (E_ / 2)];
#pragma unroll
    for (int j = 0; j < 16; ++j) {
      float4 v = xr[j];
      dst[2 * j]     = pk2bf_scaled(v.x, v.y, mk);   // bf16(x*mk) via fma_mix
      dst[2 * j + 1] = pk2bf_scaled(v.z, v.w, mk);
    }

#pragma unroll
    for (int nt2 = 0; nt2 < 2; ++nt2) {
      // B-frags of xmᵀ (stored [l][e] == Bᵀ, so row-major pattern works)
      Frag bx0 = load_frag_rowmajor(sX16, E_, 16 * nt2, 0,  lane);
      Frag bx1 = load_frag_rowmajor(sX16, E_, 16 * nt2, 32, lane);
      // H = tanh(W1 · xmᵀ); store transposed [l][d], packed dword stores
#pragma unroll
      for (int md = 0; md < 4; ++md) {
        v8f h = v8f{0.f,0.f,0.f,0.f,0.f,0.f,0.f,0.f};
        h = wmma_bf16(w1f[md][0], bx0, h);
        h = wmma_bf16(w1f[md][1], bx1, h);
        int lrow = 16 * nt2 + (lane & 15);
        int dcol = 16 * md + 8 * g;
        unsigned* hrow = (unsigned*)&sHT[wv][lrow * DA_ + dcol];
#pragma unroll
        for (int i = 0; i < 4; ++i)
          hrow[i] = pk2bf(fast_tanh(h[2 * i]), fast_tanh(h[2 * i + 1]));
      }
      // Q = W2 · H  (Hᵀ stored row-major -> row-major B pattern)
      Frag bh0 = load_frag_rowmajor(sHT[wv], DA_, 16 * nt2, 0,  lane);
      Frag bh1 = load_frag_rowmajor(sHT[wv], DA_, 16 * nt2, 32, lane);
      int lcol = 16 * nt2 + (lane & 15);
      float mkc = sMf[wv][lcol];
#pragma unroll
      for (int mr = 0; mr < 2; ++mr) {
        v8f q = v8f{0.f,0.f,0.f,0.f,0.f,0.f,0.f,0.f};
        q = wmma_bf16(w2f[mr][0], bh0, q);
        q = wmma_bf16(w2f[mr][1], bh1, q);
#pragma unroll
        for (int v = 0; v < 8; ++v) {
          float a = __expf(q[v]) * mkc;       // masked exp score
          Sacc[mr][v] += a;                   // per-lane partial row sum
          sA[wv][(16 * mr + 8 * g + v) * 32 + lcol] = f2bf(a);
        }
      }
    }

    // ---- P += A(32x32) · xm(32x64), K = 32 exactly ----
    Frag aC0 = load_frag_rowmajor(sA[wv], 32, 0,  0, lane);
    Frag aC1 = load_frag_rowmajor(sA[wv], 32, 16, 0, lane);
#pragma unroll
    for (int nt = 0; nt < 4; ++nt) {
      Frag bxc = load_fragB_colstride(sX16, E_, 0, 16 * nt, lane);
      P[0][nt] = wmma_bf16(aC0, bxc, P[0][nt]);
      P[1][nt] = wmma_bf16(aC1, bxc, P[1][nt]);
    }
  }

  // ---- reduce S across the 16 lanes of each half-wave (xor<16 stays in half) ----
#pragma unroll
  for (int mt = 0; mt < 2; ++mt)
#pragma unroll
    for (int v = 0; v < 8; ++v) {
      float s = Sacc[mt][v];
      s += __shfl_xor(s, 1, 32);
      s += __shfl_xor(s, 2, 32);
      s += __shfl_xor(s, 4, 32);
      s += __shfl_xor(s, 8, 32);
      Sacc[mt][v] = s;
    }
  if ((lane & 15) == 0) {
#pragma unroll
    for (int mt = 0; mt < 2; ++mt)
#pragma unroll
      for (int v = 0; v < 8; ++v)
        sS[wv][16 * mt + 8 * g + v] = Sacc[mt][v];
  }

  // ---- dump per-wave P partials ----
#pragma unroll
  for (int mt = 0; mt < 2; ++mt)
#pragma unroll
    for (int nt = 0; nt < 4; ++nt)
#pragma unroll
      for (int v = 0; v < 8; ++v)
        sP[wv][(16 * mt + 8 * g + v) * E_ + 16 * nt + (lane & 15)] = P[mt][nt][v];

  __syncthreads();

  // ---- cross-wave reduce + normalize + store ----
  for (int i = tid; i < R_ * E_; i += 128) {
    int r = i >> 6;
    float p = sP[0][i] + sP[1][i] + sP[2][i] + sP[3][i];
    float s = sS[0][r] + sS[1][r] + sS[2][r] + sS[3][r];
    out[(size_t)b * (R_ * E_) + i] = p / s;
  }
}

extern "C" void kernel_launch(void* const* d_in, const int* in_sizes, int n_in,
                              void* d_out, int out_size, void* d_ws, size_t ws_size,
                              hipStream_t stream) {
  const float* x    = (const float*)d_in[0];
  const float* mask = (const float*)d_in[1];
  const float* W1   = (const float*)d_in[2];
  const float* W2   = (const float*)d_in[3];
  float* out = (float*)d_out;
  (void)in_sizes; (void)n_in; (void)out_size; (void)d_ws; (void)ws_size;
  SelfBiLayer_kernel<<<B_, 128, 0, stream>>>(x, mask, W1, W2, out);
}
